// TransformerBlockQuantum_65481071399175
// MI455X (gfx1250) — matrix operand order repaired and checked by reading, not compiled, not run on hardware
//
#include <hip/hip_runtime.h>
#include <hip/hip_bf16.h>

typedef __attribute__((ext_vector_type(16))) _Float16 v16h_t;
typedef __attribute__((ext_vector_type(8)))  _Float16 v8h_t;
typedef __attribute__((ext_vector_type(8)))  float    v8f_t;

#define C_DIM 1024
#define HEADS 16
#define DK    64
#define NQ    16
#define FFN   4096
#define MT    32                     // tokens per workgroup tile (2 WMMA M-subtiles)
#define HROW_HALF (FFN + 8)          // +8 halves padding -> bank-conflict-free A reads
#define HROW_B    (HROW_HALF * 2)    // 8208 bytes per h row (16B aligned)
#define KCHUNKS   (FFN / 32)         // 128 k-chunks
#define FRAG_H    512                // halves per packed B fragment (32x16 f16 = 1KB)

// -------- kernel 0: repack w2 [C][FFN] fp32 -> B-fragment-major f16 --------
// Fragment (nTile, kChunk) is 1KB contiguous; lane L = (n&15) + 16*((k>>4)&1)
// holds K = k0..k0+15 contiguously at lane*32 bytes (contiguous k order, same as the A reads).
__global__ __launch_bounds__(256) void repack_w2_kernel(const float* __restrict__ w2,
                                                        _Float16* __restrict__ w2p) {
  int idx = blockIdx.x * 256 + threadIdx.x;  // over n*(FFN/16) + k16
  int n   = idx >> 8;                        // FFN/16 = 256
  int k   = (idx & 255) * 16;
  const float* src = w2 + (size_t)n * FFN + k;
  v8h_t h0, h1;
#pragma unroll
  for (int j = 0; j < 8; ++j) { h0[j] = (_Float16)src[j]; h1[j] = (_Float16)src[j + 8]; }
  size_t dst = ((size_t)(n >> 4) * KCHUNKS + (k >> 5)) * FRAG_H
             + (size_t)((n & 15) + 16 * ((k >> 4) & 1)) * 16;
  *(v8h_t*)(w2p + dst)     = h0;
  *(v8h_t*)(w2p + dst + 8) = h1;
}

// -------- kernel 1: per-token quantum-attn + LayerNorm -> x1, q --------
__global__ __launch_bounds__(256) void token_prep_kernel(
    const float* __restrict__ x, const float* __restrict__ gamma1,
    const float* __restrict__ beta1, float* __restrict__ x1g,
    float* __restrict__ qg) {
  __shared__ float xsh[C_DIM];
  __shared__ float csh[C_DIM];
  __shared__ float ysh[C_DIM];
  __shared__ float red[512];
  __shared__ float x1q[NQ];
  const int tid = threadIdx.x;
  const size_t t = blockIdx.x;
  const float* xrow = x + t * C_DIM;
#pragma unroll
  for (int i = 0; i < 4; ++i) {
    int c = tid + 256 * i;
    float v = xrow[c];
    xsh[c] = v;
    csh[c] = cosf(v);
  }
  __syncthreads();
  if (tid < HEADS) {                  // serial cumprod per head (64-long chain)
    int base = tid * DK;
    float cp = 1.f;
    for (int j = 0; j < DK; ++j) {
      cp *= csh[base + j];
      ysh[base + j] = xsh[base + j] + cp;   // x + attn_out
    }
  }
  __syncthreads();
  float ps = 0.f, pq = 0.f;
#pragma unroll
  for (int i = 0; i < 4; ++i) { float v = ysh[tid + 256 * i]; ps += v; pq += v * v; }
  red[tid] = ps; red[256 + tid] = pq;
  __syncthreads();
  for (int s = 128; s > 0; s >>= 1) {
    if (tid < s) { red[tid] += red[tid + s]; red[256 + tid] += red[256 + tid + s]; }
    __syncthreads();
  }
  float mu   = red[0]   * (1.f / C_DIM);
  float var  = red[256] * (1.f / C_DIM) - mu * mu;   // biased, as torch
  float rstd = rsqrtf(var + 1e-5f);
#pragma unroll
  for (int i = 0; i < 4; ++i) {
    int c = tid + 256 * i;
    float v = (ysh[c] - mu) * rstd * gamma1[c] + beta1[c];
    x1g[t * C_DIM + c] = v;
    if (c < NQ) x1q[c] = v;
  }
  __syncthreads();
  if (tid == 0) {                     // q = cumprod(cos(x1[:16]))
    float cp = 1.f;
    for (int j = 0; j < NQ; ++j) { cp *= cosf(x1q[j]); qg[t * NQ + j] = cp; }
  }
}

// -------- kernel 2: fused quantum-FFN GEMM (WMMA f16) + residual LayerNorm --------
// 32-token M-tile; ffn staging buffer aliases the h-tile LDS (dead after GEMM).
__global__ __launch_bounds__(256) void ffn_fused_kernel(
    const float* __restrict__ x1g, const float* __restrict__ qg,
    const _Float16* __restrict__ w2p, const float* __restrict__ w1,
    const float* __restrict__ b1, const float* __restrict__ b2,
    const float* __restrict__ gamma2, const float* __restrict__ beta2,
    float* __restrict__ out) {
  extern __shared__ __align__(16) char smem[];
  float* qs    = (float*)smem;                 // 32x16 f32 (2 KB)
  char*  hbase = smem + 2048;                  // 32 x 8208 B (256.5 KB)
  float* fbuf  = (float*)hbase;                // aliases h after GEMM (32x1024 f32)
  const int tid = threadIdx.x;
  const int t0  = blockIdx.x * MT;

  // phase 0: load q tile (32 tokens x 16)
  qs[tid]       = qg[(size_t)t0 * NQ + tid];
  qs[tid + 256] = qg[(size_t)t0 * NQ + tid + 256];
  __syncthreads();

  // phase 1: h[32][4096] = relu(q @ w1^T + b1) -> f16 in LDS
  for (int kk = 0; kk < FFN / 256; ++kk) {
    int k = tid + kk * 256;
    const float* w1r = w1 + (size_t)k * NQ;
    float w[NQ];
#pragma unroll
    for (int j = 0; j < NQ; ++j) w[j] = w1r[j];
    float bk = b1[k];
#pragma unroll
    for (int m = 0; m < MT; ++m) {
      float s = bk;
#pragma unroll
      for (int j = 0; j < NQ; ++j) s = fmaf(qs[m * NQ + j], w[j], s);
      s = fmaxf(s, 0.f);
      *(_Float16*)(hbase + m * HROW_B + k * 2) = (_Float16)s;
    }
  }
  __syncthreads();

  // phase 2: ffn_out = h @ w2^T via v_wmma_f32_16x16x32_f16
  const int wave = tid >> 5, lane = tid & 31;
  const int nl = lane & 15, hi = lane >> 4;
  v8f_t acc[16] = {};   // [nt] for rows 0..15, [8+nt] for rows 16..31
  // A fragment (contiguous k order, as packed B): lane<16 -> K {0..15}; lane>=16 -> {16..31}
  const char* arow0 = hbase + nl * HROW_B + hi * 32;
  const char* arow1 = arow0 + 16 * HROW_B;
  // packed-B: lane-contiguous fragment base
  const _Float16* bbase = w2p + (size_t)(wave * 8) * KCHUNKS * FRAG_H
                        + (size_t)(nl + 16 * hi) * 16;
  for (int k0 = 0; k0 < FFN; k0 += 32) {
    v8h_t a00 = *(const v8h_t*)(arow0 + k0 * 2);
    v8h_t a01 = *(const v8h_t*)(arow0 + k0 * 2 + 16);
    v8h_t a10 = *(const v8h_t*)(arow1 + k0 * 2);
    v8h_t a11 = *(const v8h_t*)(arow1 + k0 * 2 + 16);
    v16h_t a0, a1;
#pragma unroll
    for (int i = 0; i < 8; ++i) {
      a0[i] = a00[i]; a0[i + 8] = a01[i];
      a1[i] = a10[i]; a1[i + 8] = a11[i];
    }
    const _Float16* bk = bbase + (size_t)(k0 >> 5) * FRAG_H;
#pragma unroll
    for (int nt = 0; nt < 8; ++nt) {
      const _Float16* bp = bk + (size_t)nt * KCHUNKS * FRAG_H;  // 128KB immediate stride
      v8h_t b0 = *(const v8h_t*)bp;
      v8h_t b1v = *(const v8h_t*)(bp + 8);
      v16h_t b;
#pragma unroll
      for (int i = 0; i < 8; ++i) { b[i] = b0[i]; b[i + 8] = b1v[i]; }
      acc[nt]     = __builtin_amdgcn_wmma_f32_16x16x32_f16(
          false, a0, false, b, (short)0, acc[nt],     false, false);
      acc[8 + nt] = __builtin_amdgcn_wmma_f32_16x16x32_f16(
          false, a1, false, b, (short)0, acc[8 + nt], false, false);
    }
  }
  __syncthreads();   // all waves done reading h -> safe to alias fbuf over it

  // write tiles: C/D layout -> (m = v + 8*hi [+16], n = n0 + nl)
#pragma unroll
  for (int nt = 0; nt < 8; ++nt) {
    int n0 = wave * 128 + nt * 16;
#pragma unroll
    for (int v = 0; v < 8; ++v) {
      int m = v + hi * 8;
      fbuf[m * C_DIM + n0 + nl]          = acc[nt][v];
      fbuf[(m + 16) * C_DIM + n0 + nl]   = acc[8 + nt][v];
    }
  }
  __syncthreads();

  // phase 3: residual + LayerNorm (one wave per row, 4 rows per wave)
  for (int rr = 0; rr < 4; ++rr) {
    int m = wave + rr * 8;
    const float* x1row = x1g + (size_t)(t0 + m) * C_DIM;
    float vals[32];
    float ps = 0.f, pq = 0.f;
#pragma unroll
    for (int i = 0; i < 32; ++i) {
      int c = lane + i * 32;
      float v = fbuf[m * C_DIM + c] + x1row[c] + b2[c];
      vals[i] = v; ps += v; pq += v * v;
    }
#pragma unroll
    for (int off = 16; off > 0; off >>= 1) {
      ps += __shfl_xor(ps, off, 32);
      pq += __shfl_xor(pq, off, 32);
    }
    float mu   = ps * (1.f / C_DIM);
    float var  = pq * (1.f / C_DIM) - mu * mu;
    float rstd = rsqrtf(var + 1e-5f);
    float* orow = out + (size_t)(t0 + m) * C_DIM;
#pragma unroll
    for (int i = 0; i < 32; ++i) {
      int c = lane + i * 32;
      orow[c] = (vals[i] - mu) * rstd * gamma2[c] + beta2[c];
    }
  }
}

extern "C" void kernel_launch(void* const* d_in, const int* in_sizes, int n_in,
                              void* d_out, int out_size, void* d_ws, size_t ws_size,
                              hipStream_t stream) {
  (void)n_in; (void)out_size; (void)ws_size;
  const float* x      = (const float*)d_in[0];
  // d_in[1], d_in[2]: circuit phase params drop out of Z expectations -> unused
  const float* w1     = (const float*)d_in[3];
  const float* b1     = (const float*)d_in[4];
  const float* w2     = (const float*)d_in[5];
  const float* b2     = (const float*)d_in[6];
  const float* gamma1 = (const float*)d_in[7];
  const float* beta1  = (const float*)d_in[8];
  const float* gamma2 = (const float*)d_in[9];
  const float* beta2  = (const float*)d_in[10];
  float* out = (float*)d_out;

  const int tokens = in_sizes[0] / C_DIM;   // 8192

  // workspace: x1 fp32 [tokens][C] | q fp32 [tokens][16] | w2p f16 fragment-major
  char* ws = (char*)d_ws;
  float*    x1  = (float*)ws;
  float*    qb  = (float*)(ws + (size_t)tokens * C_DIM * 4);
  _Float16* w2p = (_Float16*)(ws + (size_t)tokens * C_DIM * 4 + (size_t)tokens * NQ * 4);

  repack_w2_kernel<<<(C_DIM * (FFN / 16)) / 256, 256, 0, stream>>>(w2, w2p);
  token_prep_kernel<<<tokens, 256, 0, stream>>>(x, gamma1, beta1, x1, qb);

  const size_t lds = 2048 + (size_t)MT * HROW_B;  // 264,704 B (< 320 KB WGP LDS)
  hipFuncSetAttribute((const void*)ffn_fused_kernel,
                      hipFuncAttributeMaxDynamicSharedMemorySize, (int)lds);
  ffn_fused_kernel<<<tokens / MT, 256, lds, stream>>>(x1, qb, w2p, w1, b1, b2,
                                                      gamma2, beta2, out);
}
